// InrEncoderLayer_48533130445094
// MI455X (gfx1250) — matrix operand order repaired and checked by reading, not compiled, not run on hardware
//
#include <hip/hip_runtime.h>
#include <hip/hip_bf16.h>
#include <stdint.h>

#define BB   8
#define SS   2048
#define DD   256
#define HDIM 1024
#define NEGV (-1e9f)
#define EPSV (1e-5f)

typedef __attribute__((ext_vector_type(8)))  __bf16 v8bf;
typedef __attribute__((ext_vector_type(16))) __bf16 v16bf;
typedef __attribute__((ext_vector_type(8)))  float  v8f;

#define WMMA_BF16(A, Bv, C) \
  __builtin_amdgcn_wmma_f32_16x16x32_bf16(false, (A), false, (Bv), (short)0, (C), false, false)

__device__ inline v16bf cat16(v8bf lo, v8bf hi) {
  return __builtin_shufflevector(lo, hi, 0,1,2,3,4,5,6,7,8,9,10,11,12,13,14,15);
}
// B operand per-lane chunk: 16 contiguous bf16 (32 bytes)
__device__ inline v16bf ldB(const __bf16* p) {
  v8bf lo = *(const v8bf*)p;
  v8bf hi = *(const v8bf*)(p + 8);
  return cat16(lo, hi);
}
// A operand per-lane chunk: 16 contiguous bf16 at p (K=16h..16h+15)
__device__ inline v16bf ldA(const __bf16* p) {
  v8bf lo = *(const v8bf*)p;
  v8bf hi = *(const v8bf*)(p + 8);
  return cat16(lo, hi);
}
__device__ inline v8f vzero8() {
  v8f v;
#pragma unroll
  for (int i = 0; i < 8; ++i) v[i] = 0.f;
  return v;
}
__device__ inline float wmaxf(float v) {
#pragma unroll
  for (int o = 16; o; o >>= 1) v = fmaxf(v, __shfl_xor(v, o, 32));
  return v;
}
__device__ inline float wsumf(float v) {
#pragma unroll
  for (int o = 16; o; o >>= 1) v += __shfl_xor(v, o, 32);
  return v;
}

// ---------------- weight cast + transpose (bf16, K-contiguous per output col) ----
__global__ __launch_bounds__(256) void k_prep(const float* __restrict__ w1,
                                              const float* __restrict__ w2,
                                              __bf16* __restrict__ w1T,
                                              __bf16* __restrict__ w2T) {
  int i = blockIdx.x * 256 + threadIdx.x;              // 0 .. 256*1024-1
  { int n = i >> 8,  k = i & 255;  w1T[i] = (__bf16)w1[k * HDIM + n]; }
  { int n = i >> 10, k = i & 1023; w2T[i] = (__bf16)w2[k * DD + n]; }
}

// ---------------- LN1: x -> hb (row-major bf16) and hbT ([B,D,S] bf16) ----------
__global__ __launch_bounds__(256) void k_ln1(const float* __restrict__ x,
                                             const float* __restrict__ lw,
                                             const float* __restrict__ lb,
                                             __bf16* __restrict__ hb,
                                             __bf16* __restrict__ hbT) {
  const int wave = threadIdx.x >> 5, lane = threadIdx.x & 31;
  const int row = blockIdx.x * 8 + wave;               // < B*S
  const int b = row >> 11, s = row & (SS - 1);
  const float* xr = x + (size_t)row * DD;
  float v[8]; float sum = 0.f, sq = 0.f;
#pragma unroll
  for (int j = 0; j < 8; ++j) { float t = xr[lane + 32 * j]; v[j] = t; sum += t; sq += t * t; }
  sum = wsumf(sum); sq = wsumf(sq);
  const float mu = sum * (1.f / DD);
  const float rstd = rsqrtf(sq * (1.f / DD) - mu * mu + EPSV);
#pragma unroll
  for (int j = 0; j < 8; ++j) {
    int d = lane + 32 * j;
    __bf16 h = (__bf16)((v[j] - mu) * rstd * lw[d] + lb[d]);
    hb[(size_t)row * DD + d] = h;
    hbT[((size_t)b * DD + d) * SS + s] = h;
  }
}

// ---------------- attention: y = x + softmax(QK^T/16 + softmax(masked r)) @ V ---
__global__ __launch_bounds__(256) void k_attn(const float* __restrict__ x,
                                              const float* __restrict__ r_mat,
                                              const float* __restrict__ amask,
                                              const float* __restrict__ pmask,
                                              const __bf16* __restrict__ hb,
                                              const __bf16* __restrict__ hbT,
                                              float* __restrict__ y) {
  __shared__ float smr[16][SS + 4];      // padded stride 2052 -> conflict-free
  __shared__ float outb[16][DD + 4];
  const int tid  = threadIdx.x;
  const int wave = tid >> 5, lane = tid & 31;
  const int half = lane >> 4, l15 = lane & 15;
  const int b  = blockIdx.x >> 7;
  const int q0 = (blockIdx.x & 127) << 4;

  // preload Q A-operand: 16 rows x 256, as 8 chunks of K=32 (64 VGPRs)
  v16bf qa[8];
  {
    const __bf16* qb = hb + ((size_t)b * SS + q0 + l15) * DD + half * 16;
#pragma unroll
    for (int c = 0; c < 8; ++c) qa[c] = ldA(qb + c * 32);
  }

  // Phase A: smr = softmax(masked r_mat rows); 2 rows per wave, in-wave reductions
#pragma unroll
  for (int rr = 0; rr < 2; ++rr) {
    const int m = wave * 2 + rr, q = q0 + m;
    const float* rrow = r_mat + ((size_t)b * SS + q) * SS;
    const float* arow = amask + (size_t)q * SS;
    const float* prow = pmask + (size_t)b * SS;
    float mx = NEGV;
    for (int k = lane; k < SS; k += 32) {
      float val = (prow[k] == 0.f || arow[k] == 0.f) ? NEGV : rrow[k];
      smr[m][k] = val;
      mx = fmaxf(mx, val);
    }
    mx = wmaxf(mx);
    float sm = 0.f;
    for (int k = lane; k < SS; k += 32) {
      float e = __expf(smr[m][k] - mx);
      smr[m][k] = e; sm += e;
    }
    sm = wsumf(sm);
    const float inv = 1.f / sm;
    for (int k = lane; k < SS; k += 32) smr[m][k] *= inv;
  }
  __syncthreads();

  // Phase B: scores = QK^T * (1/16) + smr, in place (128 WMMAs / wave)
  for (int kt = wave; kt < SS / 16; kt += 8) {
    const int n0 = kt << 4;
    v8f acc = vzero8();
    const __bf16* kb = hb + ((size_t)b * SS + n0 + l15) * DD + half * 16;
#pragma unroll
    for (int c = 0; c < 8; ++c) {
      v16bf bv = ldB(kb + c * 32);
      acc = WMMA_BF16(qa[c], bv, acc);
    }
    const int nn = n0 + l15;
#pragma unroll
    for (int r = 0; r < 8; ++r) {
      const int m = half * 8 + r;
      smr[m][nn] = acc[r] * 0.0625f + smr[m][nn];
    }
  }
  __syncthreads();

  // Phase C: row softmax of scores
#pragma unroll
  for (int rr = 0; rr < 2; ++rr) {
    const int m = wave * 2 + rr;
    float mx = -3.0e38f;
    for (int k = lane; k < SS; k += 32) mx = fmaxf(mx, smr[m][k]);
    mx = wmaxf(mx);
    float sm = 0.f;
    for (int k = lane; k < SS; k += 32) {
      float e = __expf(smr[m][k] - mx);
      smr[m][k] = e; sm += e;
    }
    sm = wsumf(sm);
    const float inv = 1.f / sm;
    for (int k = lane; k < SS; k += 32) smr[m][k] *= inv;
  }
  __syncthreads();

  // Phase D: O^T = V^T @ P^T ; wave owns 32 output columns (2 tiles), 128 WMMAs/wave
  {
    v8f a0 = vzero8(), a1 = vzero8();
    const __bf16* vb0 = hbT + ((size_t)b * DD + wave * 32 + l15) * SS + half * 16;
    const __bf16* vb1 = vb0 + (size_t)16 * SS;
    const float* prow = &smr[l15][half * 16];
    for (int kk = 0; kk < SS; kk += 32) {
      // B = P^T chunk: 16 contiguous f32 from prob row, cvt -> bf16
      const float4* p4 = (const float4*)(prow + kk);
      float4 f0 = p4[0], f1 = p4[1], f2 = p4[2], f3 = p4[3];
      v16bf pb;
      pb[0]=(__bf16)f0.x;  pb[1]=(__bf16)f0.y;  pb[2]=(__bf16)f0.z;  pb[3]=(__bf16)f0.w;
      pb[4]=(__bf16)f1.x;  pb[5]=(__bf16)f1.y;  pb[6]=(__bf16)f1.z;  pb[7]=(__bf16)f1.w;
      pb[8]=(__bf16)f2.x;  pb[9]=(__bf16)f2.y;  pb[10]=(__bf16)f2.z; pb[11]=(__bf16)f2.w;
      pb[12]=(__bf16)f3.x; pb[13]=(__bf16)f3.y; pb[14]=(__bf16)f3.z; pb[15]=(__bf16)f3.w;
      v16bf va0 = ldA(vb0 + kk);
      v16bf va1 = ldA(vb1 + kk);
      a0 = WMMA_BF16(va0, pb, a0);
      a1 = WMMA_BF16(va1, pb, a1);
    }
#pragma unroll
    for (int r = 0; r < 8; ++r) {
      const int nl = half * 8 + r;                 // O^T tile: M = column of O
      outb[l15][wave * 32 + nl]      = a0[r];
      outb[l15][wave * 32 + 16 + nl] = a1[r];
    }
  }
  __syncthreads();

  // residual, coalesced
#pragma unroll
  for (int m = 0; m < 16; ++m) {
    size_t g = ((size_t)b * SS + q0 + m) * DD + tid;
    y[g] = x[g] + outb[m][tid];
  }
}

// ---------------- LN2 + FFN + residual, fused per 16-row tile --------------------
__global__ __launch_bounds__(256) void k_ffn(const float* __restrict__ y,
                                             const float* __restrict__ lw,
                                             const float* __restrict__ lb,
                                             const __bf16* __restrict__ w1T,
                                             const float* __restrict__ b1,
                                             const __bf16* __restrict__ w2T,
                                             const float* __restrict__ b2,
                                             float* __restrict__ out) {
  __shared__ __bf16 h2a[16][DD + 8];     // LN2 activations (A operand, padded)
  __shared__ __bf16 tl[16][HDIM + 8];    // hidden relu activations (A operand)
  __shared__ float  ob[16][DD + 4];
  const int tid  = threadIdx.x;
  const int wave = tid >> 5, lane = tid & 31;
  const int half = lane >> 4, l15 = lane & 15;
  const int row0 = blockIdx.x << 4;

  // LN2 (2 rows / wave)
#pragma unroll
  for (int rr = 0; rr < 2; ++rr) {
    const int m = wave * 2 + rr;
    const float* yr = y + (size_t)(row0 + m) * DD;
    float v[8]; float sum = 0.f, sq = 0.f;
#pragma unroll
    for (int j = 0; j < 8; ++j) { float t = yr[lane + 32 * j]; v[j] = t; sum += t; sq += t * t; }
    sum = wsumf(sum); sq = wsumf(sq);
    const float mu = sum * (1.f / DD);
    const float rstd = rsqrtf(sq * (1.f / DD) - mu * mu + EPSV);
#pragma unroll
    for (int j = 0; j < 8; ++j) {
      int d = lane + 32 * j;
      h2a[m][d] = (__bf16)((v[j] - mu) * rstd * lw[d] + lb[d]);
    }
  }
  __syncthreads();

  // GEMM1: t = relu(h2 @ w1 + b1); wave owns 128 hidden cols (8 tiles, 64 WMMAs)
  {
    v8f acc[8];
#pragma unroll
    for (int t = 0; t < 8; ++t) acc[t] = vzero8();
    for (int kk = 0; kk < DD; kk += 32) {
      v16bf a = ldA(&h2a[l15][kk + half * 16]);
#pragma unroll
      for (int t = 0; t < 8; ++t) {
        const int n = wave * 128 + t * 16 + l15;
        v16bf bv = ldB(w1T + (size_t)n * DD + kk + half * 16);
        acc[t] = WMMA_BF16(a, bv, acc[t]);
      }
    }
#pragma unroll
    for (int t = 0; t < 8; ++t) {
      const int n = wave * 128 + t * 16 + l15;
      const float bb = b1[n];
#pragma unroll
      for (int r = 0; r < 8; ++r) {
        const int m = half * 8 + r;
        tl[m][n] = (__bf16)fmaxf(acc[t][r] + bb, 0.f);
      }
    }
  }
  __syncthreads();

  // GEMM2: o = t @ w2 + b2; wave owns 32 output cols (2 tiles, 64 WMMAs)
  {
    v8f a0 = vzero8(), a1 = vzero8();
    const int n0 = wave * 32 + l15, n1 = n0 + 16;
    for (int kk = 0; kk < HDIM; kk += 32) {
      v16bf a  = ldA(&tl[l15][kk + half * 16]);
      v16bf b0 = ldB(w2T + (size_t)n0 * HDIM + kk + half * 16);
      v16bf bv1 = ldB(w2T + (size_t)n1 * HDIM + kk + half * 16);
      a0 = WMMA_BF16(a, b0, a0);
      a1 = WMMA_BF16(a, bv1, a1);
    }
    const float bb0 = b2[n0], bb1 = b2[n1];
#pragma unroll
    for (int r = 0; r < 8; ++r) {
      const int m = half * 8 + r;
      ob[m][n0] = a0[r] + bb0;
      ob[m][n1] = a1[r] + bb1;
    }
  }
  __syncthreads();

  // residual, coalesced
#pragma unroll
  for (int m = 0; m < 16; ++m) {
    size_t g = (size_t)(row0 + m) * DD + tid;
    out[g] = y[g] + ob[m][tid];
  }
}

// ---------------- host launcher --------------------------------------------------
extern "C" void kernel_launch(void* const* d_in, const int* in_sizes, int n_in,
                              void* d_out, int out_size, void* d_ws, size_t ws_size,
                              hipStream_t stream) {
  (void)in_sizes; (void)n_in; (void)out_size; (void)ws_size;
  const float* x    = (const float*)d_in[0];
  const float* rmat = (const float*)d_in[1];
  const float* am   = (const float*)d_in[2];
  const float* pm   = (const float*)d_in[3];
  const float* l1w  = (const float*)d_in[4];
  const float* l1b  = (const float*)d_in[5];
  const float* l2w  = (const float*)d_in[6];
  const float* l2b  = (const float*)d_in[7];
  const float* w1   = (const float*)d_in[8];
  const float* b1   = (const float*)d_in[9];
  const float* w2   = (const float*)d_in[10];
  const float* b2   = (const float*)d_in[11];

  char* ws = (char*)d_ws;
  const size_t NTOK = (size_t)BB * SS;                       // 16384 rows
  __bf16* hb  = (__bf16*)ws;                                 // 8 MiB
  __bf16* hbT = (__bf16*)(ws + NTOK * DD * 2);               // 8 MiB
  float*  yb  = (float*)(ws + 2 * NTOK * DD * 2);            // 16 MiB
  __bf16* w1T = (__bf16*)(ws + 2 * NTOK * DD * 2 + NTOK * DD * 4);
  __bf16* w2T = (__bf16*)((char*)w1T + (size_t)DD * HDIM * 2);
  // total workspace used: ~33 MiB

  k_prep<<<(DD * HDIM) / 256, 256, 0, stream>>>(w1, w2, w1T, w2T);
  k_ln1 <<<NTOK / 8, 256, 0, stream>>>(x, l1w, l1b, hb, hbT);
  k_attn<<<BB * (SS / 16), 256, 0, stream>>>(x, rmat, am, pm, hb, hbT, yb);
  k_ffn <<<NTOK / 16, 256, 0, stream>>>(yb, l2w, l2b, w1T, b1, w2T, b2, (float*)d_out);
}
